// MultiHeadAttention_90031104458908
// MI455X (gfx1250) — matrix operand order repaired and checked by reading, not compiled, not run on hardware
//
#include <hip/hip_runtime.h>
#include <hip/hip_bf16.h>

typedef __attribute__((ext_vector_type(16))) __bf16 v16bf;
typedef __attribute__((ext_vector_type(8)))  float  v8f;

// ---- fragment index maps (CDNA5 16-bit WMMA layouts, CDNA5 ISA) ----
// A (16x32, 16-bit): lane -> M (lane%16); slot s -> K per half-lane swizzle.
__device__ __forceinline__ int kmapA(int s, int half) {
    return (s < 8) ? (s + 8 * half) : (s + 8 + 8 * half);
}
// B (32x16, 16-bit): lane -> N (lane%16); slot s -> K via kmapA (same as A).
// C/D (16x16 f32): vgpr r -> row r + 8*(lane/16); col = lane%16.

__device__ __forceinline__ v8f zero8() {
    v8f z;
    for (int i = 0; i < 8; ++i) z[i] = 0.0f;
    return z;
}

__device__ __forceinline__ v16bf load16bf(const __bf16* p) {
    union { v16bf v; uint4 u[2]; } t;
    t.u[0] = *(const uint4*)p;
    t.u[1] = *(const uint4*)(p + 8);
    return t.v;
}

// CDNA5 async global->LDS copy, 16B per lane, tracked by ASYNCcnt.
__device__ __forceinline__ void async_copy16(const __bf16* gptr, void* lptr) {
    unsigned lds_off = (unsigned)(uintptr_t)lptr;      // low 32 bits = LDS offset
    unsigned long long ga = (unsigned long long)(uintptr_t)gptr;
    asm volatile("global_load_async_to_lds_b128 %0, %1, off"
                 :: "v"(lds_off), "v"(ga) : "memory");
}
template <int N>
__device__ __forceinline__ void wait_async() {
    asm volatile("s_wait_asynccnt %0" :: "n"(N) : "memory");
}

// ---------------- fp32 -> bf16 convert ----------------
__global__ void cvt_f32_bf16(const float* __restrict__ in, __bf16* __restrict__ out, int n) {
    int i = blockIdx.x * blockDim.x + threadIdx.x;
    if (i < n) out[i] = (__bf16)in[i];
}

// ---------------- bf16 WMMA GEMM: C[M,N] = A[M,K] * B[K,N] ----------------
// MODE 1: bf16 head layout [B, H, T, hd]             (Q, V)
// MODE 2: fp32 row-major                             (final projection)
// MODE 3: bf16 transposed head layout [B, H, hd, T]  (K -> K^T for attention)
#define GT_SEQ 2048
#define G_NH   16
#define G_HD   128

template <int MODE>
__global__ __launch_bounds__(128)
void gemm_bf16_wmma(const __bf16* __restrict__ A, const __bf16* __restrict__ Bm,
                    void* __restrict__ Cout, int M, int N, int K)
{
    __shared__ __bf16 As[2][128][48];  // 96B row stride (16B aligned), double buffered
    __shared__ __bf16 Bs[2][32][80];   // 160B row stride (32B aligned)

    const int tid  = threadIdx.x;
    const int lane = tid & 31;
    const int wave = tid >> 5;
    const int half = lane >> 4;
    const int l16  = lane & 15;
    const int m0   = blockIdx.y * 128;
    const int n0   = blockIdx.x * 64;

    v8f acc[2][4];
    for (int i = 0; i < 2; ++i)
        for (int j = 0; j < 4; ++j) acc[i][j] = zero8();

    // async global->LDS tile staging (GLOBAL_LOAD_ASYNC_TO_LDS_B128), 6 per thread
    auto stage = [&](int buf, int k0) {
        for (int i = 0; i < 4; ++i) {
            int c = tid + i * 128;              // A: 128 rows x 4 chunks of 8 bf16
            int row = c >> 2, cc = (c & 3) * 8;
            async_copy16(&A[(size_t)(m0 + row) * K + k0 + cc], &As[buf][row][cc]);
        }
        for (int i = 0; i < 2; ++i) {
            int c = tid + i * 128;              // B: 32 rows x 8 chunks
            int row = c >> 3, cc = (c & 7) * 8;
            async_copy16(&Bm[(size_t)(k0 + row) * N + n0 + cc], &Bs[buf][row][cc]);
        }
    };

    const int nsteps = K >> 5;
    stage(0, 0);
    for (int step = 0; step < nsteps; ++step) {
        const int cur = step & 1;
        if (step + 1 < nsteps) {
            stage(cur ^ 1, (step + 1) * 32);    // prefetch next tile behind compute
            wait_async<6>();                     // in-order: tile `step` has landed
        } else {
            wait_async<0>();
        }
        __syncthreads();

        v16bf af[2];
        for (int mt = 0; mt < 2; ++mt) {
            int row = wave * 32 + mt * 16 + l16;
            for (int s = 0; s < 16; ++s) af[mt][s] = As[cur][row][kmapA(s, half)];
        }
        for (int nt = 0; nt < 4; ++nt) {
            v16bf bf;
            for (int s = 0; s < 16; ++s) bf[s] = Bs[cur][kmapA(s, half)][nt * 16 + l16];
            acc[0][nt] = __builtin_amdgcn_wmma_f32_16x16x32_bf16(
                false, af[0], false, bf, (short)0, acc[0][nt], false, false);
            acc[1][nt] = __builtin_amdgcn_wmma_f32_16x16x32_bf16(
                false, af[1], false, bf, (short)0, acc[1][nt], false, false);
        }
        __syncthreads();   // protect buffer `cur` from next iteration's prefetch
    }

    for (int mt = 0; mt < 2; ++mt)
        for (int nt = 0; nt < 4; ++nt)
            for (int r = 0; r < 8; ++r) {
                int m = m0 + wave * 32 + mt * 16 + r + 8 * half;
                int n = n0 + nt * 16 + l16;
                float v = acc[mt][nt][r];
                if constexpr (MODE == 2) {
                    ((float*)Cout)[(size_t)m * N + n] = v;
                } else if constexpr (MODE == 1) {
                    int b = m >> 11, t = m & (GT_SEQ - 1);
                    int h = n >> 7,  d = n & (G_HD - 1);
                    ((__bf16*)Cout)[(((size_t)b * G_NH + h) * GT_SEQ + t) * G_HD + d] = (__bf16)v;
                } else {   // MODE 3: K^T per head
                    int b = m >> 11, t = m & (GT_SEQ - 1);
                    int h = n >> 7,  d = n & (G_HD - 1);
                    ((__bf16*)Cout)[(((size_t)b * G_NH + h) * G_HD + d) * GT_SEQ + t] = (__bf16)v;
                }
            }
}

// ---------------- RoPE in place on [BH, T, hd] bf16 (Q) ----------------
__global__ void rope_inplace(__bf16* __restrict__ q, int BH, int T, int hd) {
    int idx = blockIdx.x * blockDim.x + threadIdx.x;
    int half = hd >> 1;
    int total = BH * T * half;
    if (idx >= total) return;
    int d  = idx % half;
    int t  = (idx / half) % T;
    int bh = idx / (half * T);
    float inv = __powf(10000.0f, -(float)(2 * d) / (float)hd);
    float ang = (float)t * inv;
    float c = __cosf(ang), s = __sinf(ang);
    size_t base = ((size_t)bh * T + t) * hd;
    float x1 = (float)q[base + d];
    float x2 = (float)q[base + d + half];
    q[base + d]        = (__bf16)(x1 * c - x2 * s);
    q[base + d + half] = (__bf16)(x2 * c + x1 * s);
}

// ---------------- RoPE in place on [BH, hd, T] bf16 (K^T) ----------------
__global__ void rope_kT(__bf16* __restrict__ kT, int BH, int T, int hd) {
    int idx = blockIdx.x * blockDim.x + threadIdx.x;
    int half = hd >> 1;
    int total = BH * T * half;
    if (idx >= total) return;
    int t  = idx % T;                       // t fastest for coalescing
    int d  = (idx / T) % half;
    int bh = idx / (T * half);
    float inv = __powf(10000.0f, -(float)(2 * d) / (float)hd);
    float ang = (float)t * inv;
    float c = __cosf(ang), s = __sinf(ang);
    size_t b1 = ((size_t)bh * hd + d) * T + t;
    size_t b2 = ((size_t)bh * hd + d + half) * T + t;
    float x1 = (float)kT[b1];
    float x2 = (float)kT[b2];
    kT[b1] = (__bf16)(x1 * c - x2 * s);
    kT[b2] = (__bf16)(x2 * c + x1 * s);
}

// ---------------- flash attention: wave = 16 query rows, hd = 128 ----------------
// K supplied transposed per head: KT[bh][d][t] -> S-fragment loads are contiguous.
__global__ __launch_bounds__(128)
void flash_attn(const __bf16* __restrict__ Q, const __bf16* __restrict__ KT,
                const __bf16* __restrict__ V, __bf16* __restrict__ Out)
{
    constexpr int T = 2048, HD = 128, NH = 16, D = 2048;
    __shared__ __bf16 Pls[4][16][40];   // per-wave P staging (C-layout -> A-layout)

    const int tid  = threadIdx.x;
    const int lane = tid & 31;
    const int wave = tid >> 5;
    const int half = lane >> 4;
    const int l16  = lane & 15;
    const int bh   = blockIdx.y;
    const int b    = bh / NH, h = bh % NH;
    const int q0   = blockIdx.x * 64 + wave * 16;
    const float scale = 0.08838834764831845f;   // 1/sqrt(128)
    const size_t baseQ = (size_t)bh * T * HD;

    // Q A-fragments: 4 slices of 16x32
    v16bf aq[4];
    for (int kk = 0; kk < 4; ++kk)
        for (int s = 0; s < 16; ++s)
            aq[kk][s] = Q[baseQ + (size_t)(q0 + l16) * HD + kk * 32 + kmapA(s, half)];

    v8f acc[8];
    for (int i = 0; i < 8; ++i) acc[i] = zero8();
    float mrow[8], lrow[8];
    for (int r = 0; r < 8; ++r) { mrow[r] = -INFINITY; lrow[r] = 0.0f; }

    const int ntiles = (q0 >> 5) + 1;           // 32-key tiles (causal bound)
    for (int j = 0; j < ntiles; ++j) {
        const int kbase = j * 32;
        if (j + 1 < ntiles) {   // hint next KV tile into cache (global_prefetch_b8)
            __builtin_prefetch(&V[baseQ + (size_t)(kbase + 32 + lane) * HD], 0, 1);
            __builtin_prefetch(&KT[baseQ + (size_t)lane * T + kbase + 32], 0, 1);
        }
        // S = Q * K^T -> two 16x16 f32 tiles; B frag: lane -> key column, slot -> d via kmapA
        v8f st[2];
        for (int t2 = 0; t2 < 2; ++t2) {
            v8f sa = zero8();
            for (int kk = 0; kk < 4; ++kk) {
                v16bf bk;
                for (int s = 0; s < 16; ++s)
                    bk[s] = KT[baseQ + (size_t)(kk * 32 + kmapA(s, half)) * T
                               + kbase + t2 * 16 + l16];
                sa = __builtin_amdgcn_wmma_f32_16x16x32_bf16(
                    false, aq[kk], false, bk, (short)0, sa, false, false);
            }
            st[t2] = sa;
        }
        // online softmax, per C-layout row (replicated across 16-lane half-group)
        for (int r = 0; r < 8; ++r) {
            const int qrow = q0 + r + 8 * half;
            float s0 = st[0][r] * scale;
            float s1 = st[1][r] * scale;
            if (kbase + l16 > qrow)      s0 = -INFINITY;
            if (kbase + 16 + l16 > qrow) s1 = -INFINITY;
            float tm = fmaxf(s0, s1);
            for (int mm = 1; mm < 16; mm <<= 1) tm = fmaxf(tm, __shfl_xor(tm, mm, 32));
            float mnew = fmaxf(mrow[r], tm);
            float corr, p0, p1;
            if (mnew == -INFINITY) { corr = 1.0f; p0 = 0.0f; p1 = 0.0f; }
            else {
                corr = __expf(mrow[r] - mnew);
                p0 = (s0 == -INFINITY) ? 0.0f : __expf(s0 - mnew);
                p1 = (s1 == -INFINITY) ? 0.0f : __expf(s1 - mnew);
            }
            float ps = p0 + p1;
            for (int mm = 1; mm < 16; mm <<= 1) ps += __shfl_xor(ps, mm, 32);
            mrow[r] = mnew;
            lrow[r] = lrow[r] * corr + ps;
            for (int nn = 0; nn < 8; ++nn) acc[nn][r] *= corr;
            const int row = r + 8 * half;
            Pls[wave][row][l16]      = (__bf16)p0;
            Pls[wave][row][16 + l16] = (__bf16)p1;
        }
        // same-wave LDS producer->consumer: DS ops are in-order within a wave;
        // only prevent compiler reordering.
        __builtin_amdgcn_wave_barrier();
        asm volatile("" ::: "memory");
        v16bf ap;   // P as 16x32 A-fragment
        for (int s = 0; s < 16; ++s)
            ap[s] = Pls[wave][l16][kmapA(s, half)];
        __builtin_amdgcn_wave_barrier();
        asm volatile("" ::: "memory");
        // O += P * V : V B-fragment: lane -> output column, slot -> key via kmapA
        for (int nn = 0; nn < 8; ++nn) {
            v16bf bv;
            for (int s = 0; s < 16; ++s)
                bv[s] = V[baseQ + (size_t)(kbase + kmapA(s, half)) * HD + nn * 16 + l16];
            acc[nn] = __builtin_amdgcn_wmma_f32_16x16x32_bf16(
                false, ap, false, bv, (short)0, acc[nn], false, false);
        }
    }

    for (int r = 0; r < 8; ++r) {
        const int row = q0 + r + 8 * half;
        float inv = (lrow[r] > 0.0f) ? 1.0f / lrow[r] : 0.0f;
        for (int nn = 0; nn < 8; ++nn) {
            float v = acc[nn][r] * inv;
            Out[((size_t)b * T + row) * D + h * HD + nn * 16 + l16] = (__bf16)v;
        }
    }
}

// ---------------- host launcher ----------------
extern "C" void kernel_launch(void* const* d_in, const int* in_sizes, int n_in,
                              void* d_out, int out_size, void* d_ws, size_t ws_size,
                              hipStream_t stream) {
    (void)in_sizes; (void)n_in; (void)out_size; (void)ws_size;
    const int Bn = 2, T = 2048, D = 2048, NH = 16, HD = 128;
    const int M = Bn * T;                      // 4096
    const size_t nXD = (size_t)M * D;          // 8M elements
    const size_t nDD = (size_t)D * D;          // 4M elements

    const float* x  = (const float*)d_in[0];
    const float* Wq = (const float*)d_in[1];
    const float* Wk = (const float*)d_in[2];
    const float* Wv = (const float*)d_in[3];
    const float* Wo = (const float*)d_in[4];

    char* ws = (char*)d_ws;
    __bf16* xb   = (__bf16*)ws;  ws += nXD * 2;
    __bf16* wqb  = (__bf16*)ws;  ws += nDD * 2;
    __bf16* wkb  = (__bf16*)ws;  ws += nDD * 2;
    __bf16* wvb  = (__bf16*)ws;  ws += nDD * 2;
    __bf16* wob  = (__bf16*)ws;  ws += nDD * 2;
    __bf16* qb   = (__bf16*)ws;  ws += nXD * 2;   // [B,H,T,hd]
    __bf16* kTb  = (__bf16*)ws;  ws += nXD * 2;   // [B,H,hd,T]
    __bf16* vb   = (__bf16*)ws;  ws += nXD * 2;   // [B,H,T,hd]
    __bf16* attn = (__bf16*)ws;  ws += nXD * 2;   // [B*T, D] row-major

    cvt_f32_bf16<<<(int)((nXD + 255) / 256), 256, 0, stream>>>(x,  xb,  (int)nXD);
    cvt_f32_bf16<<<(int)((nDD + 255) / 256), 256, 0, stream>>>(Wq, wqb, (int)nDD);
    cvt_f32_bf16<<<(int)((nDD + 255) / 256), 256, 0, stream>>>(Wk, wkb, (int)nDD);
    cvt_f32_bf16<<<(int)((nDD + 255) / 256), 256, 0, stream>>>(Wv, wvb, (int)nDD);
    cvt_f32_bf16<<<(int)((nDD + 255) / 256), 256, 0, stream>>>(Wo, wob, (int)nDD);

    dim3 gg(D / 64, M / 128);
    gemm_bf16_wmma<1><<<gg, 128, 0, stream>>>(xb, wqb, qb,  M, D, D);
    gemm_bf16_wmma<3><<<gg, 128, 0, stream>>>(xb, wkb, kTb, M, D, D);
    gemm_bf16_wmma<1><<<gg, 128, 0, stream>>>(xb, wvb, vb,  M, D, D);

    int nr = Bn * NH * T * (HD / 2);
    rope_inplace<<<(nr + 255) / 256, 256, 0, stream>>>(qb,  Bn * NH, T, HD);
    rope_kT     <<<(nr + 255) / 256, 256, 0, stream>>>(kTb, Bn * NH, T, HD);

    flash_attn<<<dim3(T / 64, Bn * NH), 128, 0, stream>>>(qb, kTb, vb, attn);

    gemm_bf16_wmma<2><<<gg, 128, 0, stream>>>(attn, wob, d_out, M, D, D);
}
